// MultiHeadSelfAttention_44229573214794
// MI455X (gfx1250) — matrix operand order repaired and checked by reading, not compiled, not run on hardware
//
#include <hip/hip_runtime.h>

typedef __attribute__((ext_vector_type(16))) __bf16 v16bf;
typedef __attribute__((ext_vector_type(8)))  float  v8f;
typedef __attribute__((ext_vector_type(4)))  unsigned int eu32x4;
typedef __attribute__((ext_vector_type(8)))  int ei32x8;
typedef __attribute__((ext_vector_type(4)))  int ei32x4;

#define EDIM  1024
#define HEADS 16
#define DKD   64
#define SEQ   2048
#define BATCH 4
#define MTOT  (BATCH * SEQ)   // 8192 rows

__device__ __forceinline__ unsigned short f2bf(float x) {
    unsigned int u = __float_as_uint(x);
    u = (u + 0x7FFFu + ((u >> 16) & 1u)) >> 16;   // round-to-nearest-even
    return (unsigned short)u;
}

union FragAB {
    v16bf bf;
    uint4 q[2];
};

__device__ __forceinline__ v8f wmma_bf16(const FragAB& a, const FragAB& b, v8f c) {
    return __builtin_amdgcn_wmma_f32_16x16x32_bf16(
        false, a.bf, false, b.bf, (short)0, c, false, false);
}

// async copy of one 16B chunk: global -> LDS (no VGPR staging, ASYNCcnt)
__device__ __forceinline__ void async_b128(const void* lds_dst, const void* gsrc) {
    unsigned ldsoff = (unsigned)(uintptr_t)lds_dst;           // low 32b = LDS offset
    unsigned long long ga = (unsigned long long)(uintptr_t)gsrc;
    asm volatile("global_load_async_to_lds_b128 %0, %1, off"
                 :: "v"(ldsoff), "v"(ga) : "memory");
}
__device__ __forceinline__ void wait_async0() {
    asm volatile("s_wait_asynccnt 0" ::: "memory");
}

// ---------------------------------------------------------------------------
// Stage 1: fp32 -> bf16 conversion
// ---------------------------------------------------------------------------
__global__ void cvt_f32_bf16(const float* __restrict__ src,
                             unsigned short* __restrict__ dst, int n) {
    int i = blockIdx.x * blockDim.x + threadIdx.x;
    if (i < n) dst[i] = f2bf(src[i]);
}

// ---------------------------------------------------------------------------
// Stage 2: fused QKV projection GEMM (grid.z selects Q/K/V)
// ---------------------------------------------------------------------------
__global__ void __launch_bounds__(256) gemm_qkv(
    const unsigned short* __restrict__ Xb,
    const unsigned short* __restrict__ Wqb,
    const unsigned short* __restrict__ Wkb,
    const unsigned short* __restrict__ Wvb,
    unsigned short* __restrict__ Qb,
    unsigned short* __restrict__ Kb,
    unsigned short* __restrict__ Vb)
{
    __shared__ unsigned short As[128][40];    // [m][k]
    __shared__ unsigned short Bts[128][40];   // [n][k] (transposed)

    const int tid  = threadIdx.x;
    const int lane = tid & 31;
    const int wave = tid >> 5;
    const int wrow = wave & 3;
    const int wcol = wave >> 2;
    const int l15  = lane & 15;
    const int lh   = lane >> 4;
    const int m0 = blockIdx.y * 128;
    const int n0 = blockIdx.x * 128;

    const unsigned short* W;
    unsigned short* Out;
    float scale;
    if (blockIdx.z == 0)      { W = Wqb; Out = Qb; scale = 0.125f; }
    else if (blockIdx.z == 1) { W = Wkb; Out = Kb; scale = 1.0f; }
    else                      { W = Wvb; Out = Vb; scale = 1.0f; }

    v8f acc[2][4];
#pragma unroll
    for (int i = 0; i < 2; ++i)
#pragma unroll
        for (int j = 0; j < 4; ++j)
            acc[i][j] = (v8f){0.f,0.f,0.f,0.f,0.f,0.f,0.f,0.f};

    for (int k0 = 0; k0 < EDIM; k0 += 32) {
        __syncthreads();
        // A tile 128x32: async global->LDS (16B per lane per op)
#pragma unroll
        for (int l = 0; l < 2; ++l) {
            int c = tid + l * 256;
            int r = c >> 2, kc = c & 3;
            async_b128(&As[r][kc * 8],
                       Xb + (size_t)(m0 + r) * EDIM + k0 + kc * 8);
        }
        // B tile 32x128, transposed into Bts[n][k]
#pragma unroll
        for (int l = 0; l < 2; ++l) {
            int c = tid + l * 256;
            int kk = c >> 4, nc = c & 15;
            uint4 v = *(const uint4*)(W + (size_t)(k0 + kk) * EDIM + n0 + nc * 8);
            const unsigned short* pv = (const unsigned short*)&v;
#pragma unroll
            for (int e = 0; e < 8; ++e) Bts[nc * 8 + e][kk] = pv[e];
        }
        if (k0 + 32 < EDIM)
            __builtin_prefetch(Xb + (size_t)(m0 + (tid >> 2)) * EDIM + k0 + 32, 0, 1);
        wait_async0();
        __syncthreads();

        FragAB aA[2], bB[4];
#pragma unroll
        for (int i = 0; i < 2; ++i) {
            int r  = wrow * 32 + i * 16 + l15;
            int kb = lh * 8;
            aA[i].q[0] = *(const uint4*)(&As[r][kb]);
            aA[i].q[1] = *(const uint4*)(&As[r][kb + 16]);
        }
#pragma unroll
        for (int j = 0; j < 4; ++j) {
            int n  = wcol * 64 + j * 16 + l15;
            int kb = lh * 8;
            bB[j].q[0] = *(const uint4*)(&Bts[n][kb]);
            bB[j].q[1] = *(const uint4*)(&Bts[n][kb + 16]);
        }
#pragma unroll
        for (int j = 0; j < 4; ++j)
#pragma unroll
            for (int i = 0; i < 2; ++i)
                acc[i][j] = wmma_bf16(aA[i], bB[j], acc[i][j]);
    }

#pragma unroll
    for (int i = 0; i < 2; ++i)
#pragma unroll
        for (int j = 0; j < 4; ++j)
#pragma unroll
            for (int r = 0; r < 8; ++r) {
                int m = m0 + wrow * 32 + i * 16 + r + lh * 8;
                int n = n0 + wcol * 64 + j * 16 + l15;
                int b = m >> 11, sidx = m & 2047;
                int h = n >> 6,  d    = n & 63;
                Out[(((size_t)b * HEADS + h) * SEQ + sidx) * DKD + d] =
                    f2bf(acc[i][j][r] * scale);
            }
}

// ---------------------------------------------------------------------------
// Stage 3: flash attention. K tile moved by the Tensor Data Mover: 2D D#
// (tile 64x64, 2B elements) with LDS padding 4 DW per 32 DW -> 144B pitch.
// ---------------------------------------------------------------------------
__global__ void __launch_bounds__(128) flash_attn(
    const unsigned short* __restrict__ Qb,
    const unsigned short* __restrict__ Kb,
    const unsigned short* __restrict__ Vb,
    unsigned short* __restrict__ Ab)
{
    __shared__ unsigned short Qs[64][72];       // [q][dk]
    __shared__ unsigned short Ks[64][72];       // [kv][dk]  (TDM destination)
    __shared__ unsigned short Vts[64][72];      // [dk][kv] (transposed)
    __shared__ unsigned short Ps[4][16][72];    // per-wave P scratch [q][kv]

    const int tid  = threadIdx.x;
    const int lane = tid & 31;
    const int wave = tid >> 5;
    const int l15  = lane & 15;
    const int lh   = lane >> 4;

    const int bh = blockIdx.y;                   // b*H + h
    const int q0 = blockIdx.x * 64;
    const size_t headOff = (size_t)bh * SEQ * DKD;
    const unsigned short* Qh = Qb + headOff;
    const unsigned short* Kh = Kb + headOff;
    const unsigned short* Vh = Vb + headOff;

    // Q tile 64x64: async global->LDS
#pragma unroll
    for (int l = 0; l < 4; ++l) {
        int c = tid + l * 128;
        int r = c >> 3, kc = c & 7;
        async_b128(&Qs[r][kc * 8], Qh + (size_t)(q0 + r) * DKD + kc * 8);
    }
    wait_async0();
    __syncthreads();

    FragAB aQ[2];
#pragma unroll
    for (int s = 0; s < 2; ++s) {
        int r  = wave * 16 + l15;
        int kb = s * 32 + lh * 8;
        aQ[s].q[0] = *(const uint4*)(&Qs[r][kb]);
        aQ[s].q[1] = *(const uint4*)(&Qs[r][kb + 16]);
    }

    v8f accO[4];
#pragma unroll
    for (int j = 0; j < 4; ++j)
        accO[j] = (v8f){0.f,0.f,0.f,0.f,0.f,0.f,0.f,0.f};
    float mrun[8], lrun[8];
#pragma unroll
    for (int r = 0; r < 8; ++r) { mrun[r] = -INFINITY; lrun[r] = 0.f; }

    // invariant parts of the K-tile tensor descriptor (D#)
    const unsigned ldsK = (unsigned)(uintptr_t)(&Ks[0][0]);

    for (int kv0 = 0; kv0 < SEQ; kv0 += 64) {
        __syncthreads();
        if (wave == 0) {
            // ---- TDM: K tile [64 x 64] bf16, global -> LDS with padding ----
            unsigned long long ga =
                (unsigned long long)(uintptr_t)(Kh + (size_t)kv0 * DKD);
            eu32x4 g0 = { 1u,                       // count=1 valid descriptor
                          ldsK,                     // lds_addr
                          (unsigned)(ga & 0xFFFFFFFFu),
                          (unsigned)((ga >> 32) & 0x1FFFFFFu) | (2u << 30) }; // type=2
            ei32x8 g1 = {
                (int)((1u << 16) |                  // data_size = 2 bytes
                      (1u << 20) |                  // pad_enable
                      (4u << 22) |                  // pad_interval: 32 DWORDs
                      (3u << 25)),                  // pad_amount:   4 DWORDs
                (int)(64u << 16),                   // tensor_dim0 = 64 (lo16)
                (int)(((unsigned)SEQ) << 16),       // dim0 hi=0 | tensor_dim1 lo16
                (int)(64u << 16),                   // dim1 hi=0 | tile_dim0 = 64
                (int)64,                            // tile_dim1 = 64 | tile_dim2 = 0
                (int)64,                            // tensor_dim0_stride = 64
                0, 0 };
            ei32x4 z4 = { 0, 0, 0, 0 };
            ei32x8 z8 = { 0, 0, 0, 0, 0, 0, 0, 0 };
            __builtin_amdgcn_tensor_load_to_lds(g0, g1, z4, z4, z8, 0);
        }
        // V tile transposed into Vts[dk][kv] (VALU path; TDM cannot transpose)
#pragma unroll
        for (int l = 0; l < 4; ++l) {
            int c = tid + l * 128;
            int r = c >> 3, kc = c & 7;
            uint4 v = *(const uint4*)(Vh + (size_t)(kv0 + r) * DKD + kc * 8);
            const unsigned short* pv = (const unsigned short*)&v;
#pragma unroll
            for (int e = 0; e < 8; ++e) Vts[kc * 8 + e][r] = pv[e];
        }
        if (wave == 0) __builtin_amdgcn_s_wait_tensorcnt(0);
        __syncthreads();

        // S = Q K^T
        v8f accS[4];
#pragma unroll
        for (int j = 0; j < 4; ++j)
            accS[j] = (v8f){0.f,0.f,0.f,0.f,0.f,0.f,0.f,0.f};
#pragma unroll
        for (int j = 0; j < 4; ++j) {
#pragma unroll
            for (int s = 0; s < 2; ++s) {
                FragAB bK;
                int n  = j * 16 + l15;
                int kk = s * 32 + lh * 8;
                bK.q[0] = *(const uint4*)(&Ks[n][kk]);
                bK.q[1] = *(const uint4*)(&Ks[n][kk + 16]);
                accS[j] = wmma_bf16(aQ[s], bK, accS[j]);
            }
        }

        // online softmax: row M lives across 16 lanes of one half-wave
        float mnew[8], alpha[8], rs[8];
#pragma unroll
        for (int r = 0; r < 8; ++r) {
            float mt = fmaxf(fmaxf(accS[0][r], accS[1][r]),
                             fmaxf(accS[2][r], accS[3][r]));
            for (int off = 8; off; off >>= 1)
                mt = fmaxf(mt, __shfl_xor(mt, off, 32));
            mnew[r]  = fmaxf(mrun[r], mt);
            alpha[r] = __expf(mrun[r] - mnew[r]);
            rs[r]    = 0.f;
        }
#pragma unroll
        for (int j = 0; j < 4; ++j)
#pragma unroll
            for (int r = 0; r < 8; ++r) {
                float p = __expf(accS[j][r] - mnew[r]);
                rs[r] += p;
                Ps[wave][r + lh * 8][j * 16 + l15] = f2bf(p);
            }
#pragma unroll
        for (int r = 0; r < 8; ++r) {
            for (int off = 8; off; off >>= 1)
                rs[r] += __shfl_xor(rs[r], off, 32);
            lrun[r] = lrun[r] * alpha[r] + rs[r];
            mrun[r] = mnew[r];
        }
#pragma unroll
        for (int j = 0; j < 4; ++j)
#pragma unroll
            for (int r = 0; r < 8; ++r)
                accO[j][r] *= alpha[r];

        __builtin_amdgcn_wave_barrier();       // wave-internal Ps ordering
        asm volatile("" ::: "memory");

        // O += P V
#pragma unroll
        for (int s = 0; s < 2; ++s) {
            FragAB aP;
            int kb = s * 32 + lh * 8;
            aP.q[0] = *(const uint4*)(&Ps[wave][l15][kb]);
            aP.q[1] = *(const uint4*)(&Ps[wave][l15][kb + 16]);
#pragma unroll
            for (int j = 0; j < 4; ++j) {
                FragAB bV;
                int n  = j * 16 + l15;
                int kk = s * 32 + lh * 8;
                bV.q[0] = *(const uint4*)(&Vts[n][kk]);
                bV.q[1] = *(const uint4*)(&Vts[n][kk + 16]);
                accO[j] = wmma_bf16(aP, bV, accO[j]);
            }
        }
    }

    // epilogue
    const int b = bh >> 4, h = bh & 15;
    float inv[8];
#pragma unroll
    for (int r = 0; r < 8; ++r) inv[r] = 1.0f / lrun[r];
#pragma unroll
    for (int j = 0; j < 4; ++j)
#pragma unroll
        for (int r = 0; r < 8; ++r) {
            int rowloc = wave * 16 + r + lh * 8;
            size_t sidx = (size_t)q0 + rowloc;
            Ab[((size_t)b * SEQ + sidx) * EDIM + h * DKD + j * 16 + l15] =
                f2bf(accO[j][r] * inv[r]);
        }
}

// ---------------------------------------------------------------------------
// Stage 4: output projection GEMM -> fp32 out
// ---------------------------------------------------------------------------
__global__ void __launch_bounds__(256) gemm_out(
    const unsigned short* __restrict__ Abuf,
    const unsigned short* __restrict__ Wob,
    float* __restrict__ Out)
{
    __shared__ unsigned short As[128][40];
    __shared__ unsigned short Bts[128][40];

    const int tid  = threadIdx.x;
    const int lane = tid & 31;
    const int wave = tid >> 5;
    const int wrow = wave & 3;
    const int wcol = wave >> 2;
    const int l15  = lane & 15;
    const int lh   = lane >> 4;
    const int m0 = blockIdx.y * 128;
    const int n0 = blockIdx.x * 128;

    v8f acc[2][4];
#pragma unroll
    for (int i = 0; i < 2; ++i)
#pragma unroll
        for (int j = 0; j < 4; ++j)
            acc[i][j] = (v8f){0.f,0.f,0.f,0.f,0.f,0.f,0.f,0.f};

    for (int k0 = 0; k0 < EDIM; k0 += 32) {
        __syncthreads();
#pragma unroll
        for (int l = 0; l < 2; ++l) {
            int c = tid + l * 256;
            int r = c >> 2, kc = c & 3;
            async_b128(&As[r][kc * 8],
                       Abuf + (size_t)(m0 + r) * EDIM + k0 + kc * 8);
        }
#pragma unroll
        for (int l = 0; l < 2; ++l) {
            int c = tid + l * 256;
            int kk = c >> 4, nc = c & 15;
            uint4 v = *(const uint4*)(Wob + (size_t)(k0 + kk) * EDIM + n0 + nc * 8);
            const unsigned short* pv = (const unsigned short*)&v;
#pragma unroll
            for (int e = 0; e < 8; ++e) Bts[nc * 8 + e][kk] = pv[e];
        }
        if (k0 + 32 < EDIM)
            __builtin_prefetch(Abuf + (size_t)(m0 + (tid >> 2)) * EDIM + k0 + 32, 0, 1);
        wait_async0();
        __syncthreads();

        FragAB aA[2], bB[4];
#pragma unroll
        for (int i = 0; i < 2; ++i) {
            int r  = wrow * 32 + i * 16 + l15;
            int kb = lh * 8;
            aA[i].q[0] = *(const uint4*)(&As[r][kb]);
            aA[i].q[1] = *(const uint4*)(&As[r][kb + 16]);
        }
#pragma unroll
        for (int j = 0; j < 4; ++j) {
            int n  = wcol * 64 + j * 16 + l15;
            int kb = lh * 8;
            bB[j].q[0] = *(const uint4*)(&Bts[n][kb]);
            bB[j].q[1] = *(const uint4*)(&Bts[n][kb + 16]);
        }
#pragma unroll
        for (int j = 0; j < 4; ++j)
#pragma unroll
            for (int i = 0; i < 2; ++i)
                acc[i][j] = wmma_bf16(aA[i], bB[j], acc[i][j]);
    }

#pragma unroll
    for (int i = 0; i < 2; ++i)
#pragma unroll
        for (int j = 0; j < 4; ++j)
#pragma unroll
            for (int r = 0; r < 8; ++r) {
                int m = m0 + wrow * 32 + i * 16 + r + lh * 8;
                int n = n0 + wcol * 64 + j * 16 + l15;
                Out[(size_t)m * EDIM + n] = acc[i][j][r];
            }
}

// ---------------------------------------------------------------------------
extern "C" void kernel_launch(void* const* d_in, const int* in_sizes, int n_in,
                              void* d_out, int out_size, void* d_ws, size_t ws_size,
                              hipStream_t stream) {
    const float* X  = (const float*)d_in[0];
    const float* Wq = (const float*)d_in[1];
    const float* Wk = (const float*)d_in[2];
    const float* Wv = (const float*)d_in[3];
    const float* Wo = (const float*)d_in[4];
    float* out = (float*)d_out;

    unsigned short* ws = (unsigned short*)d_ws;
    size_t off = 0;
    unsigned short* Xb  = ws + off; off += (size_t)MTOT * EDIM;
    unsigned short* Wqb = ws + off; off += (size_t)EDIM * EDIM;
    unsigned short* Wkb = ws + off; off += (size_t)EDIM * EDIM;
    unsigned short* Wvb = ws + off; off += (size_t)EDIM * EDIM;
    unsigned short* Wob = ws + off; off += (size_t)EDIM * EDIM;
    unsigned short* Qb  = ws + off; off += (size_t)MTOT * EDIM;
    unsigned short* Kb  = ws + off; off += (size_t)MTOT * EDIM;
    unsigned short* Vb  = ws + off; off += (size_t)MTOT * EDIM;
    unsigned short* Ab  = ws + off; off += (size_t)MTOT * EDIM;

    const int nX = MTOT * EDIM;
    const int nW = EDIM * EDIM;
    cvt_f32_bf16<<<(nX + 255) / 256, 256, 0, stream>>>(X,  Xb,  nX);
    cvt_f32_bf16<<<(nW + 255) / 256, 256, 0, stream>>>(Wq, Wqb, nW);
    cvt_f32_bf16<<<(nW + 255) / 256, 256, 0, stream>>>(Wk, Wkb, nW);
    cvt_f32_bf16<<<(nW + 255) / 256, 256, 0, stream>>>(Wv, Wvb, nW);
    cvt_f32_bf16<<<(nW + 255) / 256, 256, 0, stream>>>(Wo, Wob, nW);

    gemm_qkv<<<dim3(EDIM / 128, MTOT / 128, 3), 256, 0, stream>>>(
        Xb, Wqb, Wkb, Wvb, Qb, Kb, Vb);

    flash_attn<<<dim3(SEQ / 64, BATCH * HEADS), 128, 0, stream>>>(Qb, Kb, Vb, Ab);

    gemm_out<<<dim3(EDIM / 128, MTOT / 128), 256, 0, stream>>>(Ab, Wob, out);
}
